// point_att_60653528154666
// MI455X (gfx1250) — matrix operand order repaired and checked by reading, not compiled, not run on hardware
//
#include <hip/hip_runtime.h>
#include <hip/hip_bf16.h>
#include <stdint.h>

// ---------------------------------------------------------------------------
// Fused flash-attention for energy = X^T X, softmax, out = attn * X^T
// X = x[b] : [C=256, N=2048] fp32.  Q=K=V = X^T (N x C).
//
// Fast path (needs 16MB workspace):
//   prep kernels:  x fp32 -> bf16 copies  Xn[b][c][n]  and  Xt[b][n][c]
//   flash kernel:  bf16 WMMA, K/V tiles staged with GLOBAL_LOAD_ASYNC_TO_LDS
//                  (loop-invariant address VGPRs + immediate offsets),
//                  Q fragments in registers, f32 accum, online softmax with
//                  branch-skipped O rescale.
// Fallback path: converts in-loop (no workspace needed).
// ---------------------------------------------------------------------------

typedef __attribute__((ext_vector_type(16))) __bf16 v16bf;
typedef __attribute__((ext_vector_type(8)))  float  v8f;

union Frag16 {
  v16bf v;
  uint4 q[2];
};

#define B_      8
#define C_      256
#define N_      2048
#define KTILE   32          // keys per iteration
#define QROWS   32          // query rows per block (16 per wave)
#define THREADS 64          // 2 waves (wave32)
#define QSTRIDE 264         // 256 + 8 halves pad (row = 528B, banks spread)
#define KNSTRIDE 40         // 32 + 8 halves pad
#define PSTRIDE  40

#define WS_NEEDED ((size_t)2 * B_ * C_ * N_ * 2)   // two bf16 copies = 16 MB

__device__ __forceinline__ uint16_t f2bf(float f) {
  uint32_t u = __float_as_uint(f);
  u += 0x7FFFu + ((u >> 16) & 1u);       // round-to-nearest-even
  return (uint16_t)(u >> 16);
}

__device__ __forceinline__ float rmax16(float v) {
  v = fmaxf(v, __shfl_xor(v, 1, 32));
  v = fmaxf(v, __shfl_xor(v, 2, 32));
  v = fmaxf(v, __shfl_xor(v, 4, 32));
  v = fmaxf(v, __shfl_xor(v, 8, 32));
  return v;
}
__device__ __forceinline__ float rsum16(float v) {
  v += __shfl_xor(v, 1, 32);
  v += __shfl_xor(v, 2, 32);
  v += __shfl_xor(v, 4, 32);
  v += __shfl_xor(v, 8, 32);
  return v;
}

// low 32 bits of a generic LDS address are the LDS byte offset (flat
// addressing truncation rule), which is what async-to-LDS wants in VDST.
__device__ __forceinline__ uint32_t lds_off(const void* p) {
  return (uint32_t)(uintptr_t)p;
}

// async global->LDS, 16B per lane; OFF (imm) is added to BOTH addresses.
template <int OFF>
__device__ __forceinline__ void async_ld_b128_off(uint32_t ldsByteOff,
                                                  uint32_t gByteOff,
                                                  const void* sbase) {
  asm volatile("global_load_async_to_lds_b128 %0, %1, %2 offset:%c3"
               :: "v"(ldsByteOff), "v"(gByteOff), "s"(sbase), "i"(OFF)
               : "memory");
}
__device__ __forceinline__ void wait_asynccnt0() {
  asm volatile("s_wait_asynccnt 0x0" ::: "memory");
}

// ---------------------------------------------------------------------------
// Prep 1: natural-layout bf16 copy  Xn[b][c][n] = bf16(x[b][c][n])
// ---------------------------------------------------------------------------
__global__ __launch_bounds__(256)
void cvt_natural_kernel(const float* __restrict__ x, uint16_t* __restrict__ xn) {
  size_t i = ((size_t)blockIdx.x * 256 + threadIdx.x) * 4;
  const float4 v = *(const float4*)(x + i);
  uint32_t lo = (uint32_t)f2bf(v.x) | ((uint32_t)f2bf(v.y) << 16);
  uint32_t hi = (uint32_t)f2bf(v.z) | ((uint32_t)f2bf(v.w) << 16);
  *(uint2*)(xn + i) = make_uint2(lo, hi);
}

// ---------------------------------------------------------------------------
// Prep 2: transposed bf16 copy  Xt[b][n][c] = bf16(x[b][c][n]) via 32x32 tiles
// ---------------------------------------------------------------------------
__global__ __launch_bounds__(256)
void cvt_transpose_kernel(const float* __restrict__ x, uint16_t* __restrict__ xt) {
  __shared__ uint16_t T[32][34];     // pad to 68B rows (odd dword stride)

  const int cb = blockIdx.x & 7;                 // C_/32 = 8
  const int nb = (blockIdx.x >> 3) & 63;         // N_/32 = 64
  const int b  = blockIdx.x >> 9;
  const int c0 = cb * 32;
  const int n0 = nb * 32;
  const int t  = threadIdx.x;

  {
    const int c  = t >> 3;            // 0..31
    const int q4 = (t & 7) * 4;       // n offset in tile
    const float4 v = *(const float4*)(x + ((size_t)b * C_ + c0 + c) * N_ + n0 + q4);
    T[q4 + 0][c] = f2bf(v.x);
    T[q4 + 1][c] = f2bf(v.y);
    T[q4 + 2][c] = f2bf(v.z);
    T[q4 + 3][c] = f2bf(v.w);
  }
  __syncthreads();
  {
    const int n  = t >> 3;            // 0..31
    const int q4 = (t & 7) * 4;       // c offset in tile
    const uint2 v = *(const uint2*)&T[n][q4];
    *(uint2*)(xt + ((size_t)(b * N_ + n0 + n) * C_) + c0 + q4) = v;
  }
}

// ---------------------------------------------------------------------------
// Fast-path flash-attention kernel (bf16 inputs from workspace)
// ---------------------------------------------------------------------------
__global__ __launch_bounds__(THREADS)
void point_att_flash2_kernel(const float* __restrict__ x,
                             const uint16_t* __restrict__ xn,   // [b][c][n] bf16
                             const uint16_t* __restrict__ xt,   // [b][n][c] bf16
                             const float* __restrict__ gamma,
                             float* __restrict__ out) {
  __shared__ uint16_t Kt[KTILE][QSTRIDE];      // K tile transposed [key][channel]
  __shared__ uint16_t Kn[C_][KNSTRIDE];        // V tile natural   [channel][key]
  __shared__ uint16_t Ps[2][16][PSTRIDE];      // per-wave P staging (16 x 32)

  const int b     = blockIdx.x / (N_ / QROWS);
  const int qblk  = blockIdx.x % (N_ / QROWS);
  const int qbase = qblk * QROWS;

  const int t    = threadIdx.x;
  const int lane = t & 31;
  const int wave = t >> 5;
  const int l16  = lane & 15;
  const int hs   = lane >> 4;          // 16-lane half select

  const float*    xb  = x  + (size_t)b * C_ * N_;
  const uint16_t* xnb = xn + (size_t)b * C_ * N_;
  const uint16_t* xtb = xt + (size_t)b * N_ * C_;

  // ---- Q fragments: invariant across key loop, load once from global bf16 ----
  Frag16 qf[8];
  {
    const uint16_t* qrowg = xtb + (size_t)(qbase + wave * 16 + l16) * C_;
#pragma unroll
    for (int k = 0; k < 8; ++k) {
      qf[k].q[0] = *(const uint4*)(qrowg + 32 * k + hs * 16);
      qf[k].q[1] = *(const uint4*)(qrowg + 32 * k + hs * 16 + 8);
    }
  }

  // ---- softmax state: lane's rows are r + 8*hs (matches C/D layout) ----
  float m[8], lsum[8];
  v8f O[16];
#pragma unroll
  for (int r = 0; r < 8; ++r) { m[r] = -3.0e38f; lsum[r] = 0.0f; }
#pragma unroll
  for (int ct = 0; ct < 16; ++ct) O[ct] = (v8f){0,0,0,0,0,0,0,0};

  const uint32_t ktLds = lds_off(&Kt[0][0]);
  const uint32_t knLds = lds_off(&Kn[0][0]);

  // ---- loop-invariant async staging addresses ----
  // Kt: thread owns a fixed key-row (t&31) and segment-parity (t>>5);
  //     16 steps of 32B walk the 512B row (global and LDS strides match).
  const int rowKt = t & 31;
  const int sh    = t >> 5;
  const uint32_t ktG = (uint32_t)(rowKt * (C_ * 2) + sh * 16);
  const uint32_t ktL = ktLds + (uint32_t)(rowKt * (QSTRIDE * 2) + sh * 16);
  // Kn: thread owns 4 fixed channel-rows {t, t+64, t+128, t+192};
  //     4 x 16B segments per row (global and LDS strides match).
  uint32_t knG[4], knL[4];
#pragma unroll
  for (int g = 0; g < 4; ++g) {
    const int c = t + 64 * g;
    knG[g] = (uint32_t)(c * (N_ * 2));
    knL[g] = knLds + (uint32_t)(c * (KNSTRIDE * 2));
  }

  for (int it = 0; it < N_ / KTILE; ++it) {
    __syncthreads();                           // previous compute done
    const int j0 = it * KTILE;

    // key-tile base is block-uniform -> fold into the SGPR saddr operand
    const char* ktSB = (const char*)xtb + (size_t)j0 * (C_ * 2);
    const char* knSB = (const char*)xnb + (size_t)j0 * 2;

#define KT_LD(i) async_ld_b128_off<(i) * 32>(ktL, ktG, ktSB)
    KT_LD(0);  KT_LD(1);  KT_LD(2);  KT_LD(3);
    KT_LD(4);  KT_LD(5);  KT_LD(6);  KT_LD(7);
    KT_LD(8);  KT_LD(9);  KT_LD(10); KT_LD(11);
    KT_LD(12); KT_LD(13); KT_LD(14); KT_LD(15);
#undef KT_LD
#define KN_LD(g, s) async_ld_b128_off<(s) * 16>(knL[g], knG[g], knSB)
    KN_LD(0, 0); KN_LD(0, 1); KN_LD(0, 2); KN_LD(0, 3);
    KN_LD(1, 0); KN_LD(1, 1); KN_LD(1, 2); KN_LD(1, 3);
    KN_LD(2, 0); KN_LD(2, 1); KN_LD(2, 2); KN_LD(2, 3);
    KN_LD(3, 0); KN_LD(3, 1); KN_LD(3, 2); KN_LD(3, 3);
#undef KN_LD
    wait_asynccnt0();
    __syncthreads();

    // ---- GEMM1: S[16x32] = Q_tile x K_tile^T over C in K=32 chunks ----
    v8f S0 = (v8f){0,0,0,0,0,0,0,0};
    v8f S1 = (v8f){0,0,0,0,0,0,0,0};
#pragma unroll
    for (int k = 0; k < 8; ++k) {
      const int c0 = k * 32;
      Frag16 b0f, b1f;
      const uint16_t* k0 = &Kt[l16][c0 + hs * 16];
      b0f.q[0] = *(const uint4*)(k0);
      b0f.q[1] = *(const uint4*)(k0 + 8);
      const uint16_t* k1 = &Kt[16 + l16][c0 + hs * 16];
      b1f.q[0] = *(const uint4*)(k1);
      b1f.q[1] = *(const uint4*)(k1 + 8);
      S0 = __builtin_amdgcn_wmma_f32_16x16x32_bf16(false, qf[k].v, false, b0f.v,
                                                   (short)0, S0, false, false);
      S1 = __builtin_amdgcn_wmma_f32_16x16x32_bf16(false, qf[k].v, false, b1f.v,
                                                   (short)0, S1, false, false);
    }

    // ---- online softmax update (rows reduce across each 16-lane half) ----
    float alpha[8];
    bool changed = false;
#pragma unroll
    for (int r = 0; r < 8; ++r) {
      float rm = rmax16(fmaxf(S0[r], S1[r]));
      float mn = fmaxf(m[r], rm);
      changed |= (mn != m[r]);
      alpha[r] = __expf(m[r] - mn);
      m[r]     = mn;
    }
#pragma unroll
    for (int r = 0; r < 8; ++r) {
      S0[r] = __expf(S0[r] - m[r]);
      S1[r] = __expf(S1[r] - m[r]);
      lsum[r] = lsum[r] * alpha[r] + rsum16(S0[r] + S1[r]);
    }

    // skip the 128-fmul O rescale when no row max moved (alpha == 1 exactly)
    if (__ballot(changed)) {
#pragma unroll
      for (int ct = 0; ct < 16; ++ct)
#pragma unroll
        for (int r = 0; r < 8; ++r) O[ct][r] *= alpha[r];
    }

    // ---- convert S (C/D layout) -> P A-fragment via wave-local LDS ----
#pragma unroll
    for (int r = 0; r < 8; ++r) {
      Ps[wave][r + 8 * hs][l16]      = f2bf(S0[r]);
      Ps[wave][r + 8 * hs][16 + l16] = f2bf(S1[r]);
    }
    Frag16 p;
    const uint16_t* prow = &Ps[wave][l16][0];
    p.q[0] = *(const uint4*)(prow + hs * 16);
    p.q[1] = *(const uint4*)(prow + hs * 16 + 8);

    // ---- GEMM2: O[16x256] += P x V (K=32 over keys) ----
#pragma unroll
    for (int ct = 0; ct < 16; ++ct) {
      Frag16 vf;
      const uint16_t* vrow = &Kn[ct * 16 + l16][hs * 16];
      vf.q[0] = *(const uint4*)(vrow);
      vf.q[1] = *(const uint4*)(vrow + 8);
      O[ct] = __builtin_amdgcn_wmma_f32_16x16x32_bf16(false, p.v, false, vf.v,
                                                      (short)0, O[ct], false, false);
    }
  }

  // ---- epilogue: out = gamma * (O / l) + x ----
  const float g = gamma[0];
  const int nbase = qbase + wave * 16 + 8 * hs;
  float inv[8];
#pragma unroll
  for (int r = 0; r < 8; ++r) inv[r] = 1.0f / lsum[r];

  float* ob = out + (size_t)b * C_ * N_;
#pragma unroll
  for (int ct = 0; ct < 16; ++ct) {
    int c = ct * 16 + l16;
    const float* xp = xb + (size_t)c * N_ + nbase;
    float4 xa = *(const float4*)(xp);
    float4 xc = *(const float4*)(xp + 4);
    float4 oa, oc;
    oa.x = g * (O[ct][0] * inv[0]) + xa.x;
    oa.y = g * (O[ct][1] * inv[1]) + xa.y;
    oa.z = g * (O[ct][2] * inv[2]) + xa.z;
    oa.w = g * (O[ct][3] * inv[3]) + xa.w;
    oc.x = g * (O[ct][4] * inv[4]) + xc.x;
    oc.y = g * (O[ct][5] * inv[5]) + xc.y;
    oc.z = g * (O[ct][6] * inv[6]) + xc.z;
    oc.w = g * (O[ct][7] * inv[7]) + xc.w;
    float* op = ob + (size_t)c * N_ + nbase;
    *(float4*)(op)     = oa;
    *(float4*)(op + 4) = oc;
  }
}

// ---------------------------------------------------------------------------
// Fallback: converts fp32->bf16 in-loop (no workspace needed)
// ---------------------------------------------------------------------------
__global__ __launch_bounds__(THREADS)
void point_att_flash_kernel(const float* __restrict__ x,
                            const float* __restrict__ gamma,
                            float* __restrict__ out) {
  __shared__ uint16_t Qs[QROWS][QSTRIDE];
  __shared__ uint16_t Kt[KTILE][QSTRIDE];
  __shared__ uint16_t Kn[C_][KNSTRIDE];
  __shared__ uint16_t Ps[2][16][PSTRIDE];

  const int b     = blockIdx.x / (N_ / QROWS);
  const int qblk  = blockIdx.x % (N_ / QROWS);
  const int qbase = qblk * QROWS;

  const int t    = threadIdx.x;
  const int lane = t & 31;
  const int wave = t >> 5;
  const int l16  = lane & 15;
  const int hs   = lane >> 4;

  const float* xb = x + (size_t)b * C_ * N_;

  for (int i = 0; i < (QROWS * C_ / 4) / THREADS; ++i) {
    int idx = i * THREADS + t;
    int c   = idx >> 3;
    int q4  = (idx & 7) * 4;
    const float4 v = *(const float4*)(xb + (size_t)c * N_ + qbase + q4);
    Qs[q4 + 0][c] = f2bf(v.x);
    Qs[q4 + 1][c] = f2bf(v.y);
    Qs[q4 + 2][c] = f2bf(v.z);
    Qs[q4 + 3][c] = f2bf(v.w);
  }

  float m[8], lsum[8];
  v8f O[16];
#pragma unroll
  for (int r = 0; r < 8; ++r) { m[r] = -3.0e38f; lsum[r] = 0.0f; }
#pragma unroll
  for (int ct = 0; ct < 16; ++ct) O[ct] = (v8f){0,0,0,0,0,0,0,0};

  const uint16_t* qrow = &Qs[wave * 16 + l16][0];

  for (int it = 0; it < N_ / KTILE; ++it) {
    __syncthreads();
    const int j0 = it * KTILE;
    for (int i = 0; i < (KTILE * C_ / 4) / THREADS; ++i) {
      int idx = i * THREADS + t;
      int c   = idx >> 3;
      int j4  = (idx & 7) * 4;
      const float4 v = *(const float4*)(xb + (size_t)c * N_ + j0 + j4);
      uint16_t b0 = f2bf(v.x), b1 = f2bf(v.y), b2 = f2bf(v.z), b3 = f2bf(v.w);
      Kt[j4 + 0][c] = b0; Kt[j4 + 1][c] = b1;
      Kt[j4 + 2][c] = b2; Kt[j4 + 3][c] = b3;
      uint32_t lo = (uint32_t)b0 | ((uint32_t)b1 << 16);
      uint32_t hi = (uint32_t)b2 | ((uint32_t)b3 << 16);
      *(uint2*)&Kn[c][j4] = make_uint2(lo, hi);
    }
    __syncthreads();

    v8f S0 = (v8f){0,0,0,0,0,0,0,0};
    v8f S1 = (v8f){0,0,0,0,0,0,0,0};
#pragma unroll
    for (int c0 = 0; c0 < C_; c0 += 32) {
      Frag16 a, b0f, b1f;
      a.q[0] = *(const uint4*)(qrow + c0 + hs * 16);
      a.q[1] = *(const uint4*)(qrow + c0 + hs * 16 + 8);
      const uint16_t* k0 = &Kt[l16][c0 + hs * 16];
      b0f.q[0] = *(const uint4*)(k0);
      b0f.q[1] = *(const uint4*)(k0 + 8);
      const uint16_t* k1 = &Kt[16 + l16][c0 + hs * 16];
      b1f.q[0] = *(const uint4*)(k1);
      b1f.q[1] = *(const uint4*)(k1 + 8);
      S0 = __builtin_amdgcn_wmma_f32_16x16x32_bf16(false, a.v, false, b0f.v,
                                                   (short)0, S0, false, false);
      S1 = __builtin_amdgcn_wmma_f32_16x16x32_bf16(false, a.v, false, b1f.v,
                                                   (short)0, S1, false, false);
    }

    float alpha[8];
#pragma unroll
    for (int r = 0; r < 8; ++r) {
      float rm = rmax16(fmaxf(S0[r], S1[r]));
      float mn = fmaxf(m[r], rm);
      alpha[r] = __expf(m[r] - mn);
      m[r]     = mn;
    }
#pragma unroll
    for (int r = 0; r < 8; ++r) {
      S0[r] = __expf(S0[r] - m[r]);
      S1[r] = __expf(S1[r] - m[r]);
      lsum[r] = lsum[r] * alpha[r] + rsum16(S0[r] + S1[r]);
    }
#pragma unroll
    for (int r = 0; r < 8; ++r) {
      Ps[wave][r + 8 * hs][l16]      = f2bf(S0[r]);
      Ps[wave][r + 8 * hs][16 + l16] = f2bf(S1[r]);
    }
    Frag16 p;
    const uint16_t* prow = &Ps[wave][l16][0];
    p.q[0] = *(const uint4*)(prow + hs * 16);
    p.q[1] = *(const uint4*)(prow + hs * 16 + 8);

#pragma unroll
    for (int ct = 0; ct < 16; ++ct) {
      Frag16 vf;
      const uint16_t* vrow = &Kn[ct * 16 + l16][hs * 16];
      vf.q[0] = *(const uint4*)(vrow);
      vf.q[1] = *(const uint4*)(vrow + 8);
      v8f o = O[ct];
#pragma unroll
      for (int r = 0; r < 8; ++r) o[r] *= alpha[r];
      O[ct] = __builtin_amdgcn_wmma_f32_16x16x32_bf16(false, p.v, false, vf.v,
                                                      (short)0, o, false, false);
    }
  }

  const float g = gamma[0];
  const int nbase = qbase + wave * 16 + 8 * hs;
  float inv[8];
#pragma unroll
  for (int r = 0; r < 8; ++r) inv[r] = 1.0f / lsum[r];

  float* ob = out + (size_t)b * C_ * N_;
#pragma unroll
  for (int ct = 0; ct < 16; ++ct) {
    int c = ct * 16 + l16;
    const float* xp = xb + (size_t)c * N_ + nbase;
    float4 xa = *(const float4*)(xp);
    float4 xc = *(const float4*)(xp + 4);
    float4 oa, oc;
    oa.x = g * (O[ct][0] * inv[0]) + xa.x;
    oa.y = g * (O[ct][1] * inv[1]) + xa.y;
    oa.z = g * (O[ct][2] * inv[2]) + xa.z;
    oa.w = g * (O[ct][3] * inv[3]) + xa.w;
    oc.x = g * (O[ct][4] * inv[4]) + xc.x;
    oc.y = g * (O[ct][5] * inv[5]) + xc.y;
    oc.z = g * (O[ct][6] * inv[6]) + xc.z;
    oc.w = g * (O[ct][7] * inv[7]) + xc.w;
    float* op = ob + (size_t)c * N_ + nbase;
    *(float4*)(op)     = oa;
    *(float4*)(op + 4) = oc;
  }
}

extern "C" void kernel_launch(void* const* d_in, const int* in_sizes, int n_in,
                              void* d_out, int out_size, void* d_ws, size_t ws_size,
                              hipStream_t stream) {
  (void)in_sizes; (void)n_in; (void)out_size;
  const float* x     = (const float*)d_in[0];
  const float* gamma = (const float*)d_in[1];
  float*       out   = (float*)d_out;

  if (d_ws != nullptr && ws_size >= WS_NEEDED) {
    uint16_t* xn = (uint16_t*)d_ws;                          // [b][c][n] bf16
    uint16_t* xt = xn + (size_t)B_ * C_ * N_;                // [b][n][c] bf16

    hipLaunchKernelGGL(cvt_natural_kernel,
                       dim3((B_ * C_ * N_ / 4) / 256), dim3(256), 0, stream, x, xn);
    hipLaunchKernelGGL(cvt_transpose_kernel,
                       dim3(B_ * (N_ / 32) * (C_ / 32)), dim3(256), 0, stream, x, xt);

    hipLaunchKernelGGL(point_att_flash2_kernel,
                       dim3(B_ * (N_ / QROWS)), dim3(THREADS), 0, stream,
                       x, xn, xt, gamma, out);
  } else {
    hipLaunchKernelGGL(point_att_flash_kernel,
                       dim3(B_ * (N_ / QROWS)), dim3(THREADS), 0, stream,
                       x, gamma, out);
  }
}
